// HybridKernelRegression_65481071404325
// MI455X (gfx1250) — compile-verified, matrix operand order checked by reading, not run on hardware
//
#include <hip/hip_runtime.h>
#include <hip/hip_bf16.h>

// ---------------------------------------------------------------------------
// Kernel ridge regression (RBF) for MI455X / gfx1250.
//   K      = exp(-g * max(|a|^2+|b|^2-2 a.b, 0))  via v_wmma_f32_16x16x32_bf16
//   solve  (K + aI) w = y  via fixed-iteration CG (graph-capture safe)
//   out    = K_test @ w
// ---------------------------------------------------------------------------

typedef __attribute__((ext_vector_type(16))) __bf16 v16bf;
typedef __attribute__((ext_vector_type(8)))  __bf16 v8bf;
typedef __attribute__((ext_vector_type(8)))  float  v8f;

union Frag16 { v16bf v; v8bf h[2]; };

#define GAMMA_C 1.0f
#define ALPHA_C 1.0f
#define CG_ITERS 32

// ---------------------------------------------------------------------------
// f32 -> bf16 (RNE) conversion + row squared-norms. One block per row.
// ---------------------------------------------------------------------------
__global__ __launch_bounds__(256)
void cvt_norm_kernel(const float* __restrict__ X, unsigned short* __restrict__ Xb,
                     float* __restrict__ norms, int ncols) {
    const int row = blockIdx.x;
    const float* xr = X + (size_t)row * ncols;
    unsigned short* xbr = Xb + (size_t)row * ncols;
    float s = 0.0f;
    for (int c = threadIdx.x; c < ncols; c += blockDim.x) {
        float v = xr[c];
        s += v * v;
        unsigned int u = __float_as_uint(v);
        u += 0x7FFFu + ((u >> 16) & 1u);          // round-to-nearest-even
        xbr[c] = (unsigned short)(u >> 16);
    }
    __shared__ float red[256];
    red[threadIdx.x] = s;
    __syncthreads();
    for (int off = 128; off > 0; off >>= 1) {
        if (threadIdx.x < off) red[threadIdx.x] += red[threadIdx.x + off];
        __syncthreads();
    }
    if (threadIdx.x == 0) norms[row] = red[0];
}

// ---------------------------------------------------------------------------
// RBF Gram tile kernel, WMMA bf16 -> f32.
//   G[row, col] = exp(-g * max(na[row]+nb[col]-2*dot,0)) (+ diagAdd on diag)
// Block = 128x128 tile, 256 threads = 8 waves, each wave 32x64 (2x4 WMMA tiles).
// blockIdx.y -> rows (Xa), blockIdx.x -> cols (Xb). K multiple of 32.
// ---------------------------------------------------------------------------
__global__ __launch_bounds__(256)
void rbf_gram_wmma(const unsigned short* __restrict__ Xa,
                   const unsigned short* __restrict__ Xb,
                   const float* __restrict__ na, const float* __restrict__ nb,
                   float* __restrict__ G, int K, int ldc,
                   float gamma, float diagAdd) {
    __shared__ unsigned short As[128 * 32];
    __shared__ unsigned short Bs[128 * 32];

    const int tid  = threadIdx.x;
    const int lane = tid & 31;
    const int wave = tid >> 5;          // 0..7
    const int wr   = wave >> 1;         // 0..3 -> row offset wr*32
    const int wc   = wave & 1;          // 0..1 -> col offset wc*64
    const int blockRow = blockIdx.y * 128;
    const int blockCol = blockIdx.x * 128;

    const v8f zero8 = {0.f, 0.f, 0.f, 0.f, 0.f, 0.f, 0.f, 0.f};
    v8f acc[2][4];
    for (int i = 0; i < 2; ++i)
        for (int j = 0; j < 4; ++j)
            acc[i][j] = zero8;

    // LDS staging assignment: thread t copies 32B (16 bf16) of one row-half.
    const int copyRow  = tid >> 1;          // 0..127
    const int copyHalf = (tid & 1) * 16;    // 0 or 16 (bf16 units)

    // Fragment K-chunk base per documented 16-bit A layout (wave32):
    // lanes 0-15: K chunks {0..7, 16..23}; lanes 16-31: {8..15, 24..31}.
    const int kc = (lane < 16) ? 0 : 8;
    const int lr = lane & 15;

    for (int k0 = 0; k0 < K; k0 += 32) {
        const uint4* srcA =
            (const uint4*)(Xa + (size_t)(blockRow + copyRow) * K + k0 + copyHalf);
        const uint4* srcB =
            (const uint4*)(Xb + (size_t)(blockCol + copyRow) * K + k0 + copyHalf);
        *(uint4*)&As[copyRow * 32 + copyHalf]     = srcA[0];
        *(uint4*)&As[copyRow * 32 + copyHalf + 8] = srcA[1];
        *(uint4*)&Bs[copyRow * 32 + copyHalf]     = srcB[0];
        *(uint4*)&Bs[copyRow * 32 + copyHalf + 8] = srcB[1];
        __syncthreads();

        Frag16 af[2], bfr[4];
        for (int i = 0; i < 2; ++i) {
            const int r = wr * 32 + i * 16 + lr;
            af[i].h[0] = *(const v8bf*)&As[r * 32 + kc];
            af[i].h[1] = *(const v8bf*)&As[r * 32 + kc + 16];
        }
        for (int j = 0; j < 4; ++j) {
            const int c = wc * 64 + j * 16 + lr;
            bfr[j].h[0] = *(const v8bf*)&Bs[c * 32 + kc];
            bfr[j].h[1] = *(const v8bf*)&Bs[c * 32 + kc + 16];
        }
        for (int i = 0; i < 2; ++i)
            for (int j = 0; j < 4; ++j)
                acc[i][j] = __builtin_amdgcn_wmma_f32_16x16x32_bf16(
                    false, af[i].v, false, bfr[j].v,
                    (short)0, acc[i][j], false, false);
        __syncthreads();
    }

    // Epilogue: C layout (f32 16x16): VGPR e, lanes 0-15 -> M=e, lanes 16-31 -> M=e+8; N=lane&15
    const int n     = lane & 15;
    const int mBase = (lane >> 4) * 8;
    for (int i = 0; i < 2; ++i)
        for (int j = 0; j < 4; ++j)
            for (int e = 0; e < 8; ++e) {
                const int grow = blockRow + wr * 32 + i * 16 + mBase + e;
                const int gcol = blockCol + wc * 64 + j * 16 + n;
                float d  = acc[i][j][e];
                float sq = na[grow] + nb[gcol] - 2.0f * d;
                sq = fmaxf(sq, 0.0f);
                float g = __expf(-gamma * sq);
                if (diagAdd != 0.0f && grow == gcol) g += diagAdd;
                G[(size_t)grow * ldc + gcol] = g;
            }
}

// ---------------------------------------------------------------------------
// Dense matvec: one block per row, float4 streaming (HBM-bandwidth bound).
// ---------------------------------------------------------------------------
__global__ __launch_bounds__(256)
void matvec_kernel(const float* __restrict__ Mx, const float* __restrict__ v,
                   float* __restrict__ out, int ncols) {
    const int row = blockIdx.x;
    const float4* mr = (const float4*)(Mx + (size_t)row * ncols);
    const float4* vv = (const float4*)v;
    const int n4 = ncols >> 2;
    float s = 0.0f;
    for (int c = threadIdx.x; c < n4; c += blockDim.x) {
        float4 a = mr[c], b = vv[c];
        s += a.x * b.x + a.y * b.y + a.z * b.z + a.w * b.w;
    }
    __shared__ float red[256];
    red[threadIdx.x] = s;
    __syncthreads();
    for (int off = 128; off > 0; off >>= 1) {
        if (threadIdx.x < off) red[threadIdx.x] += red[threadIdx.x + off];
        __syncthreads();
    }
    if (threadIdx.x == 0) out[row] = red[0];
}

// ---------------------------------------------------------------------------
// Deterministic two-pass dot product.
// ---------------------------------------------------------------------------
__global__ __launch_bounds__(256)
void dot_partial_kernel(const float* __restrict__ a, const float* __restrict__ b,
                        int n, float* __restrict__ partials) {
    float s = 0.0f;
    for (int i = blockIdx.x * blockDim.x + threadIdx.x; i < n;
         i += gridDim.x * blockDim.x)
        s += a[i] * b[i];
    __shared__ float red[256];
    red[threadIdx.x] = s;
    __syncthreads();
    for (int off = 128; off > 0; off >>= 1) {
        if (threadIdx.x < off) red[threadIdx.x] += red[threadIdx.x + off];
        __syncthreads();
    }
    if (threadIdx.x == 0) partials[blockIdx.x] = red[0];
}

__global__ void reduce_partials_kernel(const float* __restrict__ partials, int n,
                                       float* __restrict__ out_scalar) {
    __shared__ float red[64];
    float s = (threadIdx.x < n) ? partials[threadIdx.x] : 0.0f;
    red[threadIdx.x] = s;
    __syncthreads();
    for (int off = 32; off > 0; off >>= 1) {
        if (threadIdx.x < off) red[threadIdx.x] += red[threadIdx.x + off];
        __syncthreads();
    }
    if (threadIdx.x == 0) out_scalar[0] = red[0];
}

// ---------------------------------------------------------------------------
// CG building blocks (scalars live in device memory -> capture safe).
// ---------------------------------------------------------------------------
__global__ void init_cg_kernel(float* __restrict__ x, float* __restrict__ r,
                               float* __restrict__ p, const float* __restrict__ y,
                               int n) {
    int i = blockIdx.x * blockDim.x + threadIdx.x;
    if (i < n) { x[i] = 0.0f; r[i] = y[i]; p[i] = y[i]; }
}

// scal[0]=rs_old, scal[1]=p.q : x += a p ; r -= a q
__global__ void cg_update1_kernel(float* __restrict__ x, float* __restrict__ r,
                                  const float* __restrict__ p,
                                  const float* __restrict__ q,
                                  const float* __restrict__ scal, int n) {
    float a = (scal[1] != 0.0f) ? (scal[0] / scal[1]) : 0.0f;
    int i = blockIdx.x * blockDim.x + threadIdx.x;
    if (i < n) { x[i] += a * p[i]; r[i] -= a * q[i]; }
}

// scal[0]=rs_old, scal[2]=rs_new : p = r + beta p
__global__ void cg_update2_kernel(float* __restrict__ p, const float* __restrict__ r,
                                  const float* __restrict__ scal, int n) {
    float beta = (scal[0] != 0.0f) ? (scal[2] / scal[0]) : 0.0f;
    int i = blockIdx.x * blockDim.x + threadIdx.x;
    if (i < n) p[i] = r[i] + beta * p[i];
}

__global__ void copy_scalar_kernel(float* __restrict__ scal) {
    scal[0] = scal[2];   // rs_old <- rs_new
}

// ---------------------------------------------------------------------------
extern "C" void kernel_launch(void* const* d_in, const int* in_sizes, int n_in,
                              void* d_out, int out_size, void* d_ws, size_t ws_size,
                              hipStream_t stream) {
    const int N = 8192;   // train rows
    const int M = 4096;   // test rows
    const int D = 512;    // feature dim

    const float* Xtr = (const float*)d_in[0];   // [N, D]
    const float* y   = (const float*)d_in[1];   // [N]
    const float* Xte = (const float*)d_in[2];   // [M, D]

    // ---- workspace carve (all chunks 256B-aligned sizes) ----
    char* w = (char*)d_ws;
    float* Kmat  = (float*)w;  w += (size_t)N * N * sizeof(float);   // 256 MB
    float* Ktest = (float*)w;  w += (size_t)M * N * sizeof(float);   // 128 MB
    unsigned short* Xa = (unsigned short*)w;  w += (size_t)N * D * sizeof(unsigned short);
    unsigned short* Xt = (unsigned short*)w;  w += (size_t)M * D * sizeof(unsigned short);
    float* na = (float*)w;  w += (size_t)N * sizeof(float);
    float* nt = (float*)w;  w += (size_t)M * sizeof(float);
    float* xv = (float*)w;  w += (size_t)N * sizeof(float);
    float* rv = (float*)w;  w += (size_t)N * sizeof(float);
    float* pv = (float*)w;  w += (size_t)N * sizeof(float);
    float* qv = (float*)w;  w += (size_t)N * sizeof(float);
    float* partials = (float*)w;  w += 64 * sizeof(float);
    float* scal     = (float*)w;  w += 16 * sizeof(float);

    // ---- bf16 conversion + squared row norms ----
    cvt_norm_kernel<<<N, 256, 0, stream>>>(Xtr, Xa, na, D);
    cvt_norm_kernel<<<M, 256, 0, stream>>>(Xte, Xt, nt, D);

    // ---- WMMA Gram matrices ----
    dim3 gTrain(N / 128, N / 128);
    rbf_gram_wmma<<<gTrain, 256, 0, stream>>>(Xa, Xa, na, na, Kmat, D, N,
                                              GAMMA_C, ALPHA_C);
    dim3 gTest(N / 128, M / 128);
    rbf_gram_wmma<<<gTest, 256, 0, stream>>>(Xt, Xa, nt, na, Ktest, D, N,
                                             GAMMA_C, 0.0f);

    // ---- CG solve (K + aI) w = y ----
    init_cg_kernel<<<N / 256, 256, 0, stream>>>(xv, rv, pv, y, N);
    dot_partial_kernel<<<64, 256, 0, stream>>>(rv, rv, N, partials);
    reduce_partials_kernel<<<1, 64, 0, stream>>>(partials, 64, &scal[0]);

    for (int it = 0; it < CG_ITERS; ++it) {
        matvec_kernel<<<N, 256, 0, stream>>>(Kmat, pv, qv, N);
        dot_partial_kernel<<<64, 256, 0, stream>>>(pv, qv, N, partials);
        reduce_partials_kernel<<<1, 64, 0, stream>>>(partials, 64, &scal[1]);
        cg_update1_kernel<<<N / 256, 256, 0, stream>>>(xv, rv, pv, qv, scal, N);
        dot_partial_kernel<<<64, 256, 0, stream>>>(rv, rv, N, partials);
        reduce_partials_kernel<<<1, 64, 0, stream>>>(partials, 64, &scal[2]);
        cg_update2_kernel<<<N / 256, 256, 0, stream>>>(pv, rv, scal, N);
        copy_scalar_kernel<<<1, 1, 0, stream>>>(scal);
    }

    // ---- prediction: out = K_test @ w ----
    matvec_kernel<<<M, 256, 0, stream>>>(Ktest, xv, (float*)d_out, N);
}
